// AFModule_26628797235715
// MI455X (gfx1250) — matrix operand order repaired and checked by reading, not compiled, not run on hardware
//
#include <hip/hip_runtime.h>
#include <hip/hip_bf16.h>

// ---------------------------------------------------------------------------
// Types for CDNA5 WMMA / TDM (gfx1250, wave32)
// ---------------------------------------------------------------------------
typedef __attribute__((ext_vector_type(16))) __bf16 v16bf;
typedef __attribute__((ext_vector_type(8)))  float  v8f;
typedef __attribute__((ext_vector_type(4)))  unsigned int u32x4;
typedef __attribute__((ext_vector_type(8)))  int          i32x8;
typedef __attribute__((ext_vector_type(4)))  int          i32x4;

union BfFrag {
    v16bf v;
    uint4 q[2];
    unsigned short u[16];
};

__device__ __forceinline__ unsigned short f2bf(float f) {
    unsigned u = __float_as_uint(f);
    unsigned r = u + 0x7FFFu + ((u >> 16) & 1u);   // round-to-nearest-even
    return (unsigned short)(r >> 16);
}

// Tensor Data Mover: 2D tile load Global->LDS using a D# descriptor in SGPRs.
// Toolchain arity differs: clang-22 (ROCm 7.2) = 5 args, clang-23 = 6 args.
__device__ __forceinline__ void tdm_load_2d(u32x4 g0, i32x8 g1) {
    i32x4 z4 = {0, 0, 0, 0};
#if defined(__clang_major__) && (__clang_major__ >= 23)
    i32x8 z8 = {0, 0, 0, 0, 0, 0, 0, 0};
    __builtin_amdgcn_tensor_load_to_lds(g0, g1, z4, z4, z8, 0);
#else
    __builtin_amdgcn_tensor_load_to_lds(g0, g1, z4, z4, 0);
#endif
}

// ---------------------------------------------------------------------------
// Generic implicit-GEMM convolution with WMMA bf16 -> f32 accumulate.
//   D[co, p] = sum_k W[co, k] * Patch[k, p]      k = c*KS*KS + kh*KS + kw
// Workgroup: 256 threads = 8 waves. Each wave: 16 couts x 64 pixels
//            = 4 x v_wmma_f32_16x16x32_bf16 per 32-wide K step.
// A tile (weights) is staged by the Tensor Data Mover (wave 0 issues the DMA,
// TDM pads each 64B row by 16B -> LDS row stride 80B == A_STR halfs).
// B tile (im2col patch) is gathered by all 256 threads; the pixel decode is
// hoisted out of the K loop (each thread owns one pixel column of the tile).
// Epilogue: out = relu?( mask[b]? * (acc*gamma[co]+beta[co]) + res? )
// ---------------------------------------------------------------------------
template<int CO, int CI, int KS, int STRIDE, int PAD,
         int HO, int WO, int HI, int WI,
         bool RELU, bool HAS_RES, bool HAS_MASK>
__global__ __launch_bounds__(256)
void conv_wmma_kernel(const unsigned short* __restrict__ Wb,   // [CO][CI*KS*KS] bf16
                      const float* __restrict__ X,             // [B][CI][HI][WI]
                      const float* __restrict__ gamma,         // [CO]
                      const float* __restrict__ beta,          // [CO]
                      const float* __restrict__ resid,         // [B][CO][HO][WO] or null
                      const float* __restrict__ mask,          // [B] or null
                      float* __restrict__ Out)                 // [B][CO][HO][WO]
{
    constexpr int KTOT    = CI * KS * KS;
    constexpr int NSTEP   = KTOT / 32;          // all layers divide exactly
    constexpr int WAVES_C = CO / 16;
    constexpr int WAVES_P = 8 / WAVES_C;
    constexpr int PIX_WG  = WAVES_P * 64;
    constexpr int A_STR   = 40;                 // halfs per A row: 64B data + 16B TDM pad
    constexpr int B_STR   = PIX_WG + 8;         // halfs per B row (16B-aligned, padded)
    constexpr int HWO     = HO * WO;
    constexpr int NIT     = (32 * PIX_WG) / 256;  // B-stage iters per thread (8 or 16)
    constexpr int KLSTEP  = 256 / PIX_WG;         // K rows advanced per iter (4 or 2)

    __shared__ __align__(16) unsigned short a_sh[2][CO * A_STR];
    __shared__ __align__(16) unsigned short b_sh[2][32 * B_STR];

    const int tid    = threadIdx.x;
    const int lane   = tid & 31;
    const int wave   = tid >> 5;
    const int wave_c = wave % WAVES_C;
    const int wave_p = wave / WAVES_C;
    const int p0     = blockIdx.x * PIX_WG;

    v8f acc[4];
#pragma unroll
    for (int t = 0; t < 4; ++t)
#pragma unroll
        for (int e = 0; e < 8; ++e) acc[t][e] = 0.0f;

    // ---- per-thread pixel decode: K-invariant, done once per kernel ----
    // 256 % PIX_WG == 0  =>  thread always serves the same pixel column.
    const int pl  = tid & (PIX_WG - 1);
    const int kl0 = tid / PIX_WG;
    int xbase, ihb, iwb;
    {
        const int p  = p0 + pl;
        const int b  = p / HWO;
        const int qq = p - b * HWO;
        const int oh = qq / WO;
        const int ow = qq - oh * WO;
        xbase = b * CI * HI * WI;
        ihb   = oh * STRIDE - PAD;
        iwb   = ow * STRIDE - PAD;
    }

    // ---- A tile via Tensor Data Mover (2D tile: CO rows x 64B, padded) ----
    auto stage_a = [&](int ks, int buf) {
        if (wave == 0) {
            const unsigned long long ga =
                (unsigned long long)(uintptr_t)(Wb + ks * 32);
            const unsigned lds =
                (unsigned)(unsigned long long)(uintptr_t)(&a_sh[buf][0]);
            u32x4 g0;
            g0.x = 1u;                                   // count=1 (valid user D#)
            g0.y = lds;                                  // lds_addr
            g0.z = (unsigned)ga;                         // global_addr[31:0]
            g0.w = (unsigned)((ga >> 32) & 0x1FFFFFFull) // global_addr[56:32]
                   | (2u << 30);                         // type = 2 ("image")
            i32x8 g1;
            g1[0] = (1 << 16)                            // data_size = 2 bytes
                  | (1 << 20)                            // pad_enable
                  | (3 << 22)                            // pad_interval: 16 DWORDs
                  | (3 << 25);                           // pad_amount:   4 DWORDs
            g1[1] = (KTOT & 0xFFFF) << 16;               // tensor_dim0[15:0]
            g1[2] = ((KTOT >> 16) & 0xFFFF)              // tensor_dim0[31:16]
                  | ((CO & 0xFFFF) << 16);               // tensor_dim1[15:0]
            g1[3] = (32 << 16);                          // tile_dim0 = 32 elems
            g1[4] = CO;                                  // tile_dim1 = CO rows
            g1[5] = KTOT;                                // tensor_dim0_stride[31:0]
            g1[6] = 0;
            g1[7] = 0;
            tdm_load_2d(g0, g1);
        }
    };

    // ---- B tile: im2col gather, only the cheap k-decode varies per step ----
    auto stage_b = [&](int ks, int buf) {
        const int k0 = ks * 32;
#pragma unroll
        for (int j = 0; j < NIT; ++j) {
            const int kl = kl0 + j * KLSTEP;
            const int k  = k0 + kl;
            const int c  = k / (KS * KS);             // const-div: mul-shift
            const int r  = k - c * (KS * KS);
            const int kh = r / KS;
            const int kw = r - kh * KS;
            const int ih = ihb + kh;
            const int iw = iwb + kw;
            float v = 0.0f;
            if (ih >= 0 && ih < HI && iw >= 0 && iw < WI)
                v = X[xbase + (c * HI + ih) * WI + iw];
            b_sh[buf][kl * B_STR + pl] = f2bf(v);
        }
    };

    stage_a(0, 0);
    stage_b(0, 0);
    __builtin_amdgcn_s_wait_tensorcnt((short)0);   // wave0's DMA done; others NOP
    __syncthreads();

    const int m    = lane & 15;
    const int half = lane >> 4;

    for (int ks = 0; ks < NSTEP; ++ks) {
        const int cur = ks & 1;
        if (ks + 1 < NSTEP) {
            stage_a(ks + 1, cur ^ 1);      // DMA overlaps this step's WMMAs
            stage_b(ks + 1, cur ^ 1);
        }

        // A fragment (16x32 bf16): lane holds row m; halves of K per lane group.
        BfFrag A;
        const unsigned short* arow = &a_sh[cur][(wave_c * 16 + m) * A_STR];
        A.q[0] = *(const uint4*)(arow + half * 8);        // K = half*8 .. +7
        A.q[1] = *(const uint4*)(arow + 16 + half * 8);   // K = 16+half*8 .. +7

#pragma unroll
        for (int t = 0; t < 4; ++t) {
            // B fragment (32x16 bf16): lane = N column (lane%16), K in table order matching A.
            BfFrag Bf;
            const unsigned short* bcol =
                &b_sh[cur][wave_p * 64 + t * 16 + m];
#pragma unroll
            for (int s = 0; s < 16; ++s)
                Bf.u[s] = bcol[((s < 8) ? (half * 8 + s) : (16 + half * 8 + (s - 8))) * B_STR];
            acc[t] = __builtin_amdgcn_wmma_f32_16x16x32_bf16(
                false, A.v, false, Bf.v, (short)0, acc[t], false, false);
        }

        if (ks + 1 < NSTEP)
            __builtin_amdgcn_s_wait_tensorcnt((short)0);  // next A slab landed
        __syncthreads();
    }

    // ---- fused epilogue: BN affine, optional mask gate, residual, ReLU ----
    float garr[8], barr[8];
#pragma unroll
    for (int r = 0; r < 8; ++r) {
        const int co = wave_c * 16 + r + 8 * half;
        garr[r] = gamma[co];
        barr[r] = beta[co];
    }
#pragma unroll
    for (int t = 0; t < 4; ++t) {
        const int p  = p0 + wave_p * 64 + t * 16 + m;     // N = lane % 16
        const int b  = p / HWO;
        const int qq = p - b * HWO;
        const float mv = HAS_MASK ? mask[b] : 1.0f;
#pragma unroll
        for (int r = 0; r < 8; ++r) {
            const int co  = wave_c * 16 + r + 8 * half;   // M = r + 8*(lane>=16)
            const int idx = (b * CO + co) * HWO + qq;
            float v = acc[t][r] * garr[r] + barr[r];
            if (HAS_MASK) v *= mv;
            if (HAS_RES)  v += resid[idx];
            if (RELU)     v = fmaxf(v, 0.0f);
            Out[idx] = v;
        }
    }
}

// ---------------------------------------------------------------------------
// Glue kernels
// ---------------------------------------------------------------------------
__global__ void cvt_bf16_kernel(const float* __restrict__ src,
                                unsigned short* __restrict__ dst, int n) {
    const int i = blockIdx.x * 256 + threadIdx.x;
    if (i < n) dst[i] = f2bf(src[i]);
}

// AvgPool2d(3, stride=2, pad=1), count_include_pad
__global__ void avgpool_kernel(const float* __restrict__ x,
                               float* __restrict__ out, int n) {
    const int i = blockIdx.x * 256 + threadIdx.x;
    if (i >= n) return;
    const int bc = i / 784, q = i - bc * 784;
    const int oh = q / 28, ow = q - oh * 28;
    const float* p = x + bc * 3136;
    float s = 0.0f;
#pragma unroll
    for (int kh = 0; kh < 3; ++kh) {
        const int ih = 2 * oh - 1 + kh;
        if (ih < 0 || ih >= 56) continue;
#pragma unroll
        for (int kw = 0; kw < 3; ++kw) {
            const int iw = 2 * ow - 1 + kw;
            if (iw >= 0 && iw < 56) s += p[ih * 56 + iw];
        }
    }
    out[i] = s * (1.0f / 9.0f);
}

// y[row] = mean over HW of x[row*HW + .]
__global__ void mean_hw_kernel(const float* __restrict__ x,
                               float* __restrict__ y, int HW) {
    __shared__ float red[256];
    const int row = blockIdx.x;
    const float* p = x + row * HW;
    float s = 0.0f;
    for (int i = threadIdx.x; i < HW; i += 256) s += p[i];
    red[threadIdx.x] = s;
    __syncthreads();
    for (int off = 128; off; off >>= 1) {
        if (threadIdx.x < off) red[threadIdx.x] += red[threadIdx.x + off];
        __syncthreads();
    }
    if (threadIdx.x == 0) y[row] = red[0] / (float)HW;
}

// navigation: 1x1 pool conv + BN + ReLU, grouped 2x2 matmul + bias,
// gumbel softmax with straight-through hard argmax (forward == hard).
__global__ void nav_kernel(const float* __restrict__ y,
                           const float* __restrict__ nav_w,
                           const float* __restrict__ bn_g,
                           const float* __restrict__ bn_b,
                           const float* __restrict__ gs_w,
                           const float* __restrict__ gs_b,
                           const float* __restrict__ gumbel,
                           float* __restrict__ mask_ws,
                           float* __restrict__ mask_out, int B) {
    const int b = blockIdx.x * blockDim.x + threadIdx.x;
    if (b >= B) return;
    const float* yb = y + b * 64;
    float g0 = 0.0f, g1 = 0.0f;
    for (int c = 0; c < 64; ++c) {
        g0 += yb[c] * nav_w[c];
        g1 += yb[c] * nav_w[64 + c];
    }
    g0 = fmaxf(g0 * bn_g[0] + bn_b[0], 0.0f);
    g1 = fmaxf(g1 * bn_g[1] + bn_b[1], 0.0f);
    const int s = b & 7;  // segment
    float z0 = g0 * gs_w[s * 4 + 0] + g1 * gs_w[s * 4 + 1] + gs_b[s * 2 + 0] + gumbel[b * 2 + 0];
    float z1 = g0 * gs_w[s * 4 + 2] + g1 * gs_w[s * 4 + 3] + gs_b[s * 2 + 1] + gumbel[b * 2 + 1];
    const float mx = fmaxf(z0, z1);
    const float e0 = __expf(z0 - mx), e1 = __expf(z1 - mx);
    const float soft1 = e1 / (e0 + e1);
    const float hard1 = (z1 > z0) ? 1.0f : 0.0f;
    const float v = (hard1 + soft1) - soft1;   // match reference fp ordering
    mask_ws[b]  = v;
    mask_out[b] = v;
}

// s[b*128+c] = mean over 56x56 of (nearest-up(xl_t) + x_big)
__global__ void fusion_mean_kernel(const float* __restrict__ xl_t,
                                   const float* __restrict__ xbig,
                                   float* __restrict__ s) {
    __shared__ float red[256];
    const int row = blockIdx.x;       // b*128 + c
    const float* xb = xbig + row * 3136;
    const float* xl = xl_t + row * 784;
    float a = 0.0f;
    for (int i = threadIdx.x; i < 3136; i += 256) {
        const int oh = i / 56, ow = i - oh * 56;
        a += xb[i] + xl[(oh >> 1) * 28 + (ow >> 1)];
    }
    red[threadIdx.x] = a;
    __syncthreads();
    for (int off = 128; off; off >>= 1) {
        if (threadIdx.x < off) red[threadIdx.x] += red[threadIdx.x + off];
        __syncthreads();
    }
    if (threadIdx.x == 0) s[row] = red[0] * (1.0f / 3136.0f);
}

// att[b,c] = sigmoid( relu(s @ w1^T) @ w2^T )   (128 -> 8 -> 128)
__global__ void att_kernel(const float* __restrict__ s,
                           const float* __restrict__ w1,
                           const float* __restrict__ w2,
                           float* __restrict__ att) {
    __shared__ float sv[128];
    __shared__ float h[8];
    const int b = blockIdx.x, t = threadIdx.x;
    sv[t] = s[b * 128 + t];
    __syncthreads();
    if (t < 8) {
        float a = 0.0f;
        for (int c = 0; c < 128; ++c) a += sv[c] * w1[t * 128 + c];
        h[t] = fmaxf(a, 0.0f);
    }
    __syncthreads();
    float a = 0.0f;
#pragma unroll
    for (int j = 0; j < 8; ++j) a += h[j] * w2[t * 8 + j];
    att[b * 128 + t] = 1.0f / (1.0f + __expf(-a));
}

// out0 = relu( att*up(xl) + (1-att)*x_big )
__global__ void blend_kernel(const float* __restrict__ xl_t,
                             const float* __restrict__ xbig,
                             const float* __restrict__ att,
                             float* __restrict__ out0, int n) {
    const int i = blockIdx.x * 256 + threadIdx.x;
    if (i >= n) return;
    const int row = i / 3136, q = i - row * 3136;
    const int oh = q / 56, ow = q - oh * 56;
    const float xl = xl_t[row * 784 + (oh >> 1) * 28 + (ow >> 1)];
    const float a  = att[row];
    out0[i] = fmaxf(a * xl + (1.0f - a) * xbig[i], 0.0f);
}

// ---------------------------------------------------------------------------
// Host-side orchestration
// ---------------------------------------------------------------------------
extern "C" void kernel_launch(void* const* d_in, const int* in_sizes, int n_in,
                              void* d_out, int out_size, void* d_ws, size_t ws_size,
                              hipStream_t stream) {
    (void)in_sizes; (void)n_in; (void)out_size; (void)ws_size;

    const float* x        = (const float*)d_in[0];
    const float* gumbel   = (const float*)d_in[1];
    const float* nav_w    = (const float*)d_in[2];
    const float* nav_bn_g = (const float*)d_in[3];
    const float* nav_bn_b = (const float*)d_in[4];
    const float* gs_w     = (const float*)d_in[5];
    const float* gs_b     = (const float*)d_in[6];
    const float* ba_w1    = (const float*)d_in[7];
    const float* ba_g1    = (const float*)d_in[8];
    const float* ba_b1    = (const float*)d_in[9];
    const float* ba_w2    = (const float*)d_in[10];
    const float* ba_g2    = (const float*)d_in[11];
    const float* ba_b2    = (const float*)d_in[12];
    const float* bf_wd    = (const float*)d_in[13];
    const float* bf_gd    = (const float*)d_in[14];
    const float* bf_bd    = (const float*)d_in[15];
    const float* bf_w1    = (const float*)d_in[16];
    const float* bf_g1    = (const float*)d_in[17];
    const float* bf_b1    = (const float*)d_in[18];
    const float* bf_w2    = (const float*)d_in[19];
    const float* bf_g2    = (const float*)d_in[20];
    const float* bf_b2    = (const float*)d_in[21];
    const float* bt_w     = (const float*)d_in[22];
    const float* bt_g     = (const float*)d_in[23];
    const float* bt_b     = (const float*)d_in[24];
    const float* att_w1   = (const float*)d_in[25];
    const float* att_w2   = (const float*)d_in[26];
    const float* fu_w1    = (const float*)d_in[27];
    const float* fu_g1    = (const float*)d_in[28];
    const float* fu_b1    = (const float*)d_in[29];
    const float* fu_w2    = (const float*)d_in[30];
    const float* fu_g2    = (const float*)d_in[31];
    const float* fu_b2    = (const float*)d_in[32];

    constexpr int B = 128;
    constexpr int N28 = B * 64 * 784;          // 6,422,528
    constexpr int N56 = B * 128 * 3136;        // 51,380,224

    // ---- workspace carve (floats), with reuse of the big 56x56 buffers ----
    float* fw = (float*)d_ws;
    size_t o = 0;
    auto alloc = [&](size_t n) { float* p = fw + o; o += n; return p; };
    float* y      = alloc(B * 64);
    float* maskw  = alloc(B);
    float* svec   = alloc(B * 128);
    float* attw   = alloc(B * 128);
    float* res_l  = alloc(N28);
    float* h28    = alloc(N28);
    float* xlit   = alloc(N28);
    float* xlt    = alloc(B * 128 * 784);
    float* res_b  = alloc(N56);
    float* h56    = alloc(N56);
    float* xbig   = alloc(N56);
    float* out0   = res_b;   // dead after x_big
    float* h56b   = h56;     // dead after x_big

    unsigned short* wb = (unsigned short*)(fw + o);
    size_t wo = 0;
    auto walloc = [&](size_t n) { unsigned short* p = wb + wo; wo += n; return p; };
    unsigned short* wb_ba1 = walloc(64 * 576);
    unsigned short* wb_ba2 = walloc(64 * 576);
    unsigned short* wb_fd  = walloc(128 * 64);
    unsigned short* wb_f1  = walloc(128 * 576);
    unsigned short* wb_f2  = walloc(128 * 1152);
    unsigned short* wb_bt  = walloc(128 * 64);
    unsigned short* wb_fu1 = walloc(128 * 1152);
    unsigned short* wb_fu2 = walloc(128 * 1152);

    float* out      = (float*)d_out;
    float* mask_out = out + N56;

    auto cvt = [&](const float* src, unsigned short* dst, int n) {
        cvt_bf16_kernel<<<(n + 255) / 256, 256, 0, stream>>>(src, dst, n);
    };
    cvt(ba_w1, wb_ba1, 64 * 576);
    cvt(ba_w2, wb_ba2, 64 * 576);
    cvt(bf_wd, wb_fd, 128 * 64);
    cvt(bf_w1, wb_f1, 128 * 576);
    cvt(bf_w2, wb_f2, 128 * 1152);
    cvt(bt_w, wb_bt, 128 * 64);
    cvt(fu_w1, wb_fu1, 128 * 1152);
    cvt(fu_w2, wb_fu2, 128 * 1152);

    // navigation mask
    mean_hw_kernel<<<B * 64, 256, 0, stream>>>(x, y, 3136);
    nav_kernel<<<1, 128, 0, stream>>>(y, nav_w, nav_bn_g, nav_bn_b,
                                      gs_w, gs_b, gumbel, maskw, mask_out, B);

    // ample branch (28x28)
    avgpool_kernel<<<(N28 + 255) / 256, 256, 0, stream>>>(x, res_l, N28);
    conv_wmma_kernel<64, 64, 3, 2, 1, 28, 28, 56, 56, true, false, false>
        <<<B * 784 / 128, 256, 0, stream>>>(wb_ba1, x, ba_g1, ba_b1, nullptr, nullptr, h28);
    conv_wmma_kernel<64, 64, 3, 1, 1, 28, 28, 28, 28, false, true, false>
        <<<B * 784 / 128, 256, 0, stream>>>(wb_ba2, h28, ba_g2, ba_b2, res_l, nullptr, xlit);
    conv_wmma_kernel<128, 64, 1, 1, 0, 28, 28, 28, 28, false, false, false>
        <<<B * 784 / 64, 256, 0, stream>>>(wb_bt, xlit, bt_g, bt_b, nullptr, nullptr, xlt);

    // focal branch (56x56), conv2 gated by per-segment mask
    conv_wmma_kernel<128, 64, 1, 1, 0, 56, 56, 56, 56, false, false, false>
        <<<B * 3136 / 64, 256, 0, stream>>>(wb_fd, x, bf_gd, bf_bd, nullptr, nullptr, res_b);
    conv_wmma_kernel<128, 64, 3, 1, 1, 56, 56, 56, 56, true, false, false>
        <<<B * 3136 / 64, 256, 0, stream>>>(wb_f1, x, bf_g1, bf_b1, nullptr, nullptr, h56);
    conv_wmma_kernel<128, 128, 3, 1, 1, 56, 56, 56, 56, false, true, true>
        <<<B * 3136 / 64, 256, 0, stream>>>(wb_f2, h56, bf_g2, bf_b2, res_b, maskw, xbig);

    // SE-style dynamic fusion
    fusion_mean_kernel<<<B * 128, 256, 0, stream>>>(xlt, xbig, svec);
    att_kernel<<<B, 128, 0, stream>>>(svec, att_w1, att_w2, attw);
    blend_kernel<<<(N56 + 255) / 256, 256, 0, stream>>>(xlt, xbig, attw, out0, N56);

    // fusion residual block (last_relu = true), final conv writes d_out
    conv_wmma_kernel<128, 128, 3, 1, 1, 56, 56, 56, 56, true, false, false>
        <<<B * 3136 / 64, 256, 0, stream>>>(wb_fu1, out0, fu_g1, fu_b1, nullptr, nullptr, h56b);
    conv_wmma_kernel<128, 128, 3, 1, 1, 56, 56, 56, 56, true, true, false>
        <<<B * 3136 / 64, 256, 0, stream>>>(wb_fu2, h56b, fu_g2, fu_b2, out0, nullptr, out);
}
